// ASA_80942953660434
// MI455X (gfx1250) — matrix operand order repaired and checked by reading, not compiled, not run on hardware
//
#include <hip/hip_runtime.h>
#include <hip/hip_bf16.h>
#include <cstdint>

// Problem constants from the reference: B=4, C=256, H=W=64 -> HW=4096, N=M=4096.
#define C_   256
#define HW_  4096
#define BC   128          // m-columns processed per attention block iteration

typedef __attribute__((ext_vector_type(16))) _Float16 v16h;
typedef __attribute__((ext_vector_type(8)))  float    v8f;

struct U4 { uint32_t x, y, z, w; };   // POD 16B vector for b128 loads/stores
struct F2 { float x, y; };

union Frag16 {                        // one WMMA f16 A/B fragment (8 VGPRs)
    v16h      v;
    _Float16  h[16];
    uint32_t  u[8];
};

#define WMMA_F16(A, B, Cc) \
    __builtin_amdgcn_wmma_f32_16x16x32_f16(false, (A), false, (B), (short)0, (Cc), false, false)

// Packed-K layout: for batch b, c-chunk kc (32 channels), m-tile mt (16 cols):
//   kpk[((b*8+kc)*(HW/16)+mt)*32*16 + lane*16 + e]
// holds element (c = kc*32 + (lane>>4)*16 + e, m = mt*16 + (lane&15)) so an
// attention B-fragment is exactly lane-contiguous 32 bytes (2 x b128 loads).

// ---------------------------------------------------------------------------
// 1x1 conv as WMMA GEMM: out[co,p] = sum_ci W[co,ci] * X[ci,p] + bias[co]
//   mode 0: write transposed f16  out0[b][p][co]               (Q)
//   mode 1: write B-fragment-packed f16 (see above)            (K)
//   mode 2: write natural [C][M] f16, out0 = V, out1 = V*V     (V, V^2)
// ---------------------------------------------------------------------------
__global__ __launch_bounds__(256) void proj_kernel(
    const float* __restrict__ X,     // [B][C][HW] f32
    const float* __restrict__ W,     // [C][C]     f32
    const float* __restrict__ bias,  // [C]        f32
    _Float16* __restrict__ out0,
    _Float16* __restrict__ out1,
    int mode)
{
    const int tid  = threadIdx.x;
    const int wave = tid >> 5;
    const int lane = tid & 31;
    const int lr   = lane & 15;      // row/col within 16
    const int hi   = lane >> 4;      // half-wave select

    const int pt = blockIdx.x % (HW_ / 16);
    const int b  = blockIdx.x / (HW_ / 16);
    const int p0 = pt * 16;

    // --- Load 8 B-fragments of X (K = ci, 32 per chunk); col p = p0+lr ---
    Frag16 xb[8];
#pragma unroll
    for (int kc = 0; kc < 8; ++kc) {
        const int cr = kc * 32 + hi * 16;
        const float* xp = X + ((size_t)b * C_ + cr) * HW_ + p0 + lr;
#pragma unroll
        for (int v = 0; v < 8; ++v) {
            xb[kc].h[2 * v]     = (_Float16)xp[(size_t)(2 * v) * HW_];
            xb[kc].h[2 * v + 1] = (_Float16)xp[(size_t)(2 * v + 1) * HW_];
        }
    }

#pragma unroll
    for (int t = 0; t < 2; ++t) {                 // two co-tiles per wave
        const int co0 = wave * 32 + t * 16;
        v8f acc = {0.f, 0.f, 0.f, 0.f, 0.f, 0.f, 0.f, 0.f};
#pragma unroll
        for (int kc = 0; kc < 8; ++kc) {
            // A-fragment of W: row co = co0+lr, K = ci (contiguous pairs)
            Frag16 wa;
            const float* wp = W + (size_t)(co0 + lr) * C_ + kc * 32 + hi * 16;
#pragma unroll
            for (int j = 0; j < 4; ++j) {
                F2 f = *(const F2*)(wp + 2 * j);
                wa.h[2 * j]     = (_Float16)f.x;
                wa.h[2 * j + 1] = (_Float16)f.y;
            }
#pragma unroll
            for (int j = 0; j < 4; ++j) {
                F2 f = *(const F2*)(wp + 8 + 2 * j);
                wa.h[8 + 2 * j]     = (_Float16)f.x;
                wa.h[8 + 2 * j + 1] = (_Float16)f.y;
            }
            acc = WMMA_F16(wa.v, xb[kc].v, acc);
        }
        // D layout: lane col p = p0+lr, row co = co0 + r + 8*hi
        if (mode == 0) {
            // Q: contiguous 8 channels per lane -> one b128 store
            union { U4 u4; _Float16 h[8]; } pk;
#pragma unroll
            for (int r = 0; r < 8; ++r)
                pk.h[r] = (_Float16)(acc[r] + bias[co0 + r + 8 * hi]);
            *(U4*)(out0 + ((size_t)b * HW_ + p0 + lr) * C_ + co0 + 8 * hi) = pk.u4;
        } else if (mode == 1) {
            // K packed: this D tile's rows are one half-lane-group's fragment
            // bytes. kc = wave (co0>>5), u = t, dest lane = t*16 + lr.
            union { U4 u4; _Float16 h[8]; } pk;
#pragma unroll
            for (int r = 0; r < 8; ++r)
                pk.h[r] = (_Float16)(acc[r] + bias[co0 + r + 8 * hi]);
            _Float16* dst = out0 +
                (((size_t)b * 8 + wave) * (HW_ / 16) + pt) * (32 * 16) +
                (t * 16 + lr) * 16 + hi * 8;
            *(U4*)dst = pk.u4;
        } else {
#pragma unroll
            for (int r = 0; r < 8; ++r) {
                const int co = co0 + r + 8 * hi;
                const float d = acc[r] + bias[co];
                const size_t idx = ((size_t)b * C_ + co) * HW_ + p0 + lr;
                out0[idx] = (_Float16)d;
                out1[idx] = (_Float16)(d * d);
            }
        }
    }
}

// ---------------------------------------------------------------------------
// Per-(b,c) mean and 1/sqrt(var+eps) of content, unbiased var (ddof=1).
// ---------------------------------------------------------------------------
__global__ __launch_bounds__(256) void stats_kernel(
    const float* __restrict__ x, float* __restrict__ cmean, float* __restrict__ cinv)
{
    __shared__ float ss[256], sq[256];
    const int g = blockIdx.x;                 // b*C + c
    const float* p = x + (size_t)g * HW_;
    float s = 0.f, s2 = 0.f;
    for (int i = threadIdx.x; i < HW_; i += 256) { float v = p[i]; s += v; s2 += v * v; }
    ss[threadIdx.x] = s; sq[threadIdx.x] = s2;
    __syncthreads();
    for (int off = 128; off > 0; off >>= 1) {
        if (threadIdx.x < off) {
            ss[threadIdx.x] += ss[threadIdx.x + off];
            sq[threadIdx.x] += sq[threadIdx.x + off];
        }
        __syncthreads();
    }
    if (threadIdx.x == 0) {
        const float m   = ss[0] / (float)HW_;
        const float var = (sq[0] - (float)HW_ * m * m) / (float)(HW_ - 1);
        cmean[g] = m;
        cinv[g]  = rsqrtf(var + 1e-5f);
    }
}

// ---------------------------------------------------------------------------
// Fused flash-attention: softmax(QK^T) against V and V^2, online softmax,
// std/mean epilogue fused with mean-variance-norm of content.
// One workgroup = one (batch, 16-row n-tile). 8 waves; wave w owns output
// channels [32w, 32w+32). PV computed as O^T = V^T * S^T.
// ---------------------------------------------------------------------------
__global__ __launch_bounds__(256) void attn_kernel(
    const _Float16* __restrict__ q16,   // [B][N][C]
    const _Float16* __restrict__ kpk,   // packed-K (see header comment)
    const _Float16* __restrict__ v16,   // [B][C][M]
    const _Float16* __restrict__ v2q,   // [B][C][M]
    const float* __restrict__ cmean,
    const float* __restrict__ cinv,
    const float* __restrict__ content,  // [B][C][HW]
    float* __restrict__ out)            // [B][C][HW]
{
    __shared__ __align__(16) float    Sraw[16 * BC];    // [n][m] raw logits
    __shared__ __align__(16) _Float16 Spk[4 * 32 * 16]; // packed S^T fragments
    __shared__ float wmax[8 * 16];                      // per-wave row maxima
    __shared__ float mrow[16], lrow[16], srow[16], lpart[16];

    const int tid  = threadIdx.x;
    const int wave = tid >> 5;
    const int lane = tid & 31;
    const int lr   = lane & 15;
    const int hi   = lane >> 4;

    const int b  = blockIdx.x / (HW_ / 16);
    const int n0 = (blockIdx.x % (HW_ / 16)) * 16;

    if (tid < 16) { mrow[tid] = -1e30f; lrow[tid] = 0.f; }

    // --- Q A-fragments for this n-tile: row n = n0+lr, K = c (8 chunks) ---
    Frag16 qf[8];
    {
        const _Float16* qrow = q16 + ((size_t)b * HW_ + n0 + lr) * C_;
        const int base = hi * 16;
#pragma unroll
        for (int kc = 0; kc < 8; ++kc) {
            *(U4*)&qf[kc].u[0] = *(const U4*)(qrow + kc * 32 + base);
            *(U4*)&qf[kc].u[4] = *(const U4*)(qrow + kc * 32 + 8 + base);
        }
    }

    v8f accv[2] = { {0.f,0.f,0.f,0.f,0.f,0.f,0.f,0.f}, {0.f,0.f,0.f,0.f,0.f,0.f,0.f,0.f} };
    v8f acc2[2] = { {0.f,0.f,0.f,0.f,0.f,0.f,0.f,0.f}, {0.f,0.f,0.f,0.f,0.f,0.f,0.f,0.f} };

    const size_t kbatch = (size_t)b * 8 * (HW_ / 16);   // kc-major tile index base

    for (int m0 = 0; m0 < HW_; m0 += BC) {
        __syncthreads();   // protect LDS reuse from previous iteration

        // ---- 1. Each wave: one 16x16 S tile at m-cols [m0+16*wave, +16) ----
        {
            const int mt = (m0 >> 4) + wave;            // m-tile index
            v8f s = {0.f,0.f,0.f,0.f,0.f,0.f,0.f,0.f};
#pragma unroll
            for (int kc = 0; kc < 8; ++kc) {
                Frag16 bf_;
                const _Float16* p =
                    kpk + ((kbatch + (size_t)kc * (HW_ / 16) + mt) * 32 + lane) * 16;
                *(U4*)&bf_.u[0] = *(const U4*)p;
                *(U4*)&bf_.u[4] = *(const U4*)(p + 8);
                s = WMMA_F16(qf[kc].v, bf_.v, s);
            }
            // prefetch next block's K for this wave (global_prefetch_b8)
            if (m0 + BC < HW_) {
                const _Float16* pn =
                    kpk + ((kbatch + ((m0 + BC) >> 4) + wave) * 32 + lane) * 16;
                __builtin_prefetch(pn, 0, 1);
            }
            // D layout: lane col m = wave*16+lr, rows n = r + 8*hi
#pragma unroll
            for (int r = 0; r < 8; ++r)
                Sraw[(r + 8 * hi) * BC + wave * 16 + lr] = s[r];

            // In-wave row max: reduce each row across its 16 columns
            float m8[8];
#pragma unroll
            for (int r = 0; r < 8; ++r) m8[r] = s[r];
#pragma unroll
            for (int x = 1; x < 16; x <<= 1)
#pragma unroll
                for (int r = 0; r < 8; ++r)
                    m8[r] = fmaxf(m8[r], __shfl_xor(m8[r], x, 32));
            if (lr == 0) {
#pragma unroll
                for (int r = 0; r < 8; ++r)
                    wmax[wave * 16 + 8 * hi + r] = m8[r];
            }
        }
        __syncthreads();

        // ---- 2. Running-max update / accumulator scale (wave 0) ----
        if (wave == 0 && lane < 16) {
            float bmax = wmax[lane];
#pragma unroll
            for (int w = 1; w < 8; ++w) bmax = fmaxf(bmax, wmax[w * 16 + lane]);
            const float mo = mrow[lane];
            const float mn = fmaxf(mo, bmax);
            mrow[lane]  = mn;
            srow[lane]  = __expf(mo - mn);
            lpart[lane] = 0.f;
        }
        __syncthreads();

        // ---- 3. exp -> packed S^T fragments (f16) + partial row sums ----
        {
            const int n  = tid & 15;
            const int mb = (tid >> 4) * 8;              // 8 consecutive m
            const float mx = mrow[n];
            float psum = 0.f;
            union { U4 u4; _Float16 h[8]; } pk;
#pragma unroll
            for (int i = 0; i < 8; ++i) {
                const float e = __expf(Sraw[n * BC + mb + i] - mx);
                psum += e;
                pk.h[i] = (_Float16)e;
            }
            const int mc = mb >> 5;                     // 32-m chunk
            const int w  = mb & 31;
            *(U4*)&Spk[((mc * 32) + ((w & 8) << 1) + n) * 16 + ((w & 16) >> 1)] = pk.u4;
            atomicAdd(&lpart[n], psum);
        }
        __syncthreads();

        if (wave == 0 && lane < 16)
            lrow[lane] = lrow[lane] * srow[lane] + lpart[lane];

        // ---- 4. Rescale accumulators (per output column n = lr) ----
        {
            const float sc = srow[lr];
#pragma unroll
            for (int t = 0; t < 2; ++t)
#pragma unroll
                for (int r = 0; r < 8; ++r) { accv[t][r] *= sc; acc2[t][r] *= sc; }
        }

        // ---- 5. O^T += V^T(32c x 32m) * S^T(32m x 16n), 4 m-chunks ----
#pragma unroll
        for (int mc = 0; mc < 4; ++mc) {
            Frag16 sb;                    // B-fragment of S^T from packed LDS
            const _Float16* sp = &Spk[(mc * 32 + lane) * 16];
            *(U4*)&sb.u[0] = *(const U4*)sp;
            *(U4*)&sb.u[4] = *(const U4*)(sp + 8);
#pragma unroll
            for (int t = 0; t < 2; ++t) {
                const int c0 = wave * 32 + t * 16;
                const size_t rowoff = ((size_t)b * C_ + c0 + lr) * HW_ + m0 + mc * 32;
                Frag16 va;                // A-fragment of V^T: row c, K = m
                *(U4*)&va.u[0] = *(const U4*)(v16 + rowoff + hi * 8);
                *(U4*)&va.u[4] = *(const U4*)(v16 + rowoff + 16 + hi * 8);
                accv[t] = WMMA_F16(va.v, sb.v, accv[t]);
                Frag16 v2a;
                *(U4*)&v2a.u[0] = *(const U4*)(v2q + rowoff + hi * 8);
                *(U4*)&v2a.u[4] = *(const U4*)(v2q + rowoff + 16 + hi * 8);
                acc2[t] = WMMA_F16(v2a.v, sb.v, acc2[t]);
            }
        }
    }
    __syncthreads();

    // ---- Epilogue: mean/std + fused mean-variance-norm of content ----
    {
        const float linv = 1.0f / lrow[lr];     // softmax denom for col n
        const int n = n0 + lr;
#pragma unroll
        for (int t = 0; t < 2; ++t) {
#pragma unroll
            for (int r = 0; r < 8; ++r) {
                const int c = wave * 32 + t * 16 + r + 8 * hi;
                const float mean = accv[t][r] * linv;
                const float e2   = acc2[t][r] * linv;
                const float sd   = sqrtf(fmaxf(e2 - mean * mean, 0.f));
                const size_t idx = ((size_t)b * C_ + c) * HW_ + n;
                const int gc = b * C_ + c;
                const float normed = (content[idx] - cmean[gc]) * cinv[gc];
                out[idx] = sd * normed + mean;
            }
        }
    }
}

// ---------------------------------------------------------------------------
extern "C" void kernel_launch(void* const* d_in, const int* in_sizes, int n_in,
                              void* d_out, int out_size, void* d_ws, size_t ws_size,
                              hipStream_t stream)
{
    const float* content = (const float*)d_in[0];
    const float* style   = (const float*)d_in[1];
    const float* ckey    = (const float*)d_in[2];
    const float* skey    = (const float*)d_in[3];
    const float* Wf = (const float*)d_in[4]; const float* bf = (const float*)d_in[5];
    const float* Wg = (const float*)d_in[6]; const float* bg = (const float*)d_in[7];
    const float* Wh = (const float*)d_in[8]; const float* bh = (const float*)d_in[9];
    float* out = (float*)d_out;

    const int B = in_sizes[0] / (C_ * HW_);     // = 4

    const size_t nHalf = (size_t)B * HW_ * C_;  // elements per f16 tensor
    _Float16* q16 = (_Float16*)d_ws;
    _Float16* kpk = q16 + nHalf;
    _Float16* v16 = kpk + nHalf;
    _Float16* v2q = v16 + nHalf;
    float* cm   = (float*)(v2q + nHalf);
    float* ci   = cm + (size_t)B * C_;

    const dim3 blk(256);
    const dim3 gproj(B * (HW_ / 16));
    proj_kernel<<<gproj, blk, 0, stream>>>(ckey, Wf, bf, q16, nullptr, 0);
    proj_kernel<<<gproj, blk, 0, stream>>>(skey, Wg, bg, kpk, nullptr, 1);
    proj_kernel<<<gproj, blk, 0, stream>>>(style, Wh, bh, v16, v2q, 2);
    stats_kernel<<<dim3(B * C_), blk, 0, stream>>>(content, cm, ci);
    attn_kernel<<<gproj, blk, 0, stream>>>(q16, kpk, v16, v2q, cm, ci, content, out);
}
